// Res_Auto_3d_Model_Sparse_Occu_111669149978
// MI455X (gfx1250) — matrix operand order repaired and checked by reading, not compiled, not run on hardware
//
#include <hip/hip_runtime.h>
#include <hip/hip_fp16.h>
#include <stdint.h>
#include <stddef.h>

// ---------------------------------------------------------------------------
// CDNA5 WMMA (wave32): A/B = v16h (8 VGPRs packed f16), C/D = v8f.
// A-fragment per lane = two contiguous 8-half chunks: [8g,8g+8) and [16+8g,+8)
// B-fragment per lane = the same two 8-half K chunks of column n (if W is n-major)
// ---------------------------------------------------------------------------
typedef _Float16 half_t;
typedef __attribute__((ext_vector_type(16))) _Float16 v16h;
typedef __attribute__((ext_vector_type(8)))  _Float16 v8h;
typedef __attribute__((ext_vector_type(8)))  float    v8f;

#define NB_ 64           // batch
#define NS_ 32           // grid side
#define NQ_ 512          // queries per sample
#define MDEC (NB_ * NQ_) // 32768 decoder rows

__device__ __forceinline__ float selu_f(float x) {
  const float a = 1.6732632423543772f, s = 1.0507009873554805f;
  return x > 0.f ? s * x : s * a * (__expf(x) - 1.f);
}
__device__ __forceinline__ float sigmoid_f(float x) { return 1.f / (1.f + __expf(-x)); }

// scattered A K-index (only used by the tiny-ci scalar fallback)
__device__ __forceinline__ int a_k_base(int g, int v) {
  return (v < 4) ? (8 * g + 2 * v) : (16 + 8 * g + 2 * (v - 4));
}

// ---------------------------------------------------------------------------
// Utility kernels
// ---------------------------------------------------------------------------
__global__ void zero_f32_kernel(float* p, size_t n) {
  size_t i = (size_t)blockIdx.x * blockDim.x + threadIdx.x;
  if (i < n) p[i] = 0.f;
}
__global__ void zero_f16_kernel(half_t* p, size_t n) {
  size_t i = (size_t)blockIdx.x * blockDim.x + threadIdx.x;
  if (i < n) p[i] = (half_t)0.f;
}

__global__ void voxelize_kernel(const float* __restrict__ feats,
                                const int* __restrict__ coors,
                                const int* __restrict__ bidx,
                                float* __restrict__ ssum, float* __restrict__ cnt, int n) {
  int i = blockIdx.x * blockDim.x + threadIdx.x;
  if (i >= n) return;
  int b = bidx[i];
  int c0 = coors[3 * i], c1 = coors[3 * i + 1], c2 = coors[3 * i + 2];
  size_t lin = (((size_t)b * NS_ + c0) * NS_ + c1) * NS_ + c2;
  atomicAdd(&ssum[lin], feats[i]);
  atomicAdd(&cnt[lin], 1.f);
}

__global__ void finalize_voxel_kernel(const float* __restrict__ ssum,
                                      const float* __restrict__ cnt,
                                      half_t* __restrict__ x0, float* __restrict__ mask,
                                      size_t total) {
  size_t i = (size_t)blockIdx.x * blockDim.x + threadIdx.x;
  if (i >= total) return;
  float c = cnt[i];
  x0[i] = (half_t)(ssum[i] / fmaxf(c, 1.f));
  mask[i] = (c > 0.f) ? 1.f : 0.f;
}

// f32 [K][N] row-major -> f16 TRANSPOSED zero-padded [Np][Kp] (n-major, K contiguous)
__global__ void pack_wT_kernel(const float* __restrict__ src, half_t* __restrict__ dst,
                               int K, int N, int Kp, int Np) {
  int i = blockIdx.x * blockDim.x + threadIdx.x;
  if (i >= Kp * Np) return;
  int n = i / Kp, k = i % Kp;
  dst[i] = (half_t)((k < K && n < N) ? src[k * N + n] : 0.f);
}

// conv weights f32 [taps][ci][co] -> f16 [taps][Np][32] (per-tap n-major, K padded to 32)
__global__ void pack_conv_kernel(const float* __restrict__ src, half_t* __restrict__ dst,
                                 int ntaps, int ci, int co, int Np) {
  int i = blockIdx.x * blockDim.x + threadIdx.x;
  if (i >= ntaps * Np * 32) return;
  int t = i / (Np * 32);
  int rem = i % (Np * 32);
  int n = rem / 32, k = rem % 32;
  dst[i] = (half_t)((k < ci && n < co) ? src[(t * ci + k) * co + n] : 0.f);
}

__global__ void pool_x_kernel(const half_t* __restrict__ xin, half_t* __restrict__ xout,
                              int Rout, int c, size_t total) {
  size_t i = (size_t)blockIdx.x * blockDim.x + threadIdx.x;
  if (i >= total) return;
  int ch = (int)(i % c);
  size_t v = i / c;
  int R3o = Rout * Rout * Rout;
  int b = (int)(v / R3o);
  int vv = (int)(v % R3o);
  int z = vv / (Rout * Rout), y = (vv / Rout) % Rout, x = vv % Rout;
  int Rin = Rout * 2;
  float m = -3.0e38f;
  for (int dz = 0; dz < 2; ++dz)
    for (int dy = 0; dy < 2; ++dy)
      for (int dx = 0; dx < 2; ++dx) {
        size_t idx = (((size_t)b * Rin * Rin * Rin) +
                      (size_t)(2 * z + dz) * Rin * Rin + (2 * y + dy) * Rin + (2 * x + dx)) * c + ch;
        m = fmaxf(m, (float)xin[idx]);
      }
  xout[i] = (half_t)m;
}

__global__ void pool_mask_kernel(const float* __restrict__ min_, float* __restrict__ mout,
                                 int Rout, size_t total) {
  size_t i = (size_t)blockIdx.x * blockDim.x + threadIdx.x;
  if (i >= total) return;
  int R3o = Rout * Rout * Rout;
  int b = (int)(i / R3o);
  int vv = (int)(i % R3o);
  int z = vv / (Rout * Rout), y = (vv / Rout) % Rout, x = vv % Rout;
  int Rin = Rout * 2;
  float m = 0.f;
  for (int dz = 0; dz < 2; ++dz)
    for (int dy = 0; dy < 2; ++dy)
      for (int dx = 0; dx < 2; ++dx) {
        size_t idx = ((size_t)b * Rin * Rin * Rin) +
                     (size_t)(2 * z + dz) * Rin * Rin + (2 * y + dy) * Rin + (2 * x + dx);
        m = fmaxf(m, min_[idx]);
      }
  mout[i] = m;
}

// q0: [MDEC][96] f16, cols 0..63 = z[b], 64..66 = query xyz, 67..95 = 0
__global__ void build_q_kernel(const half_t* __restrict__ z, const float* __restrict__ coors_de,
                               half_t* __restrict__ q, size_t total) {
  size_t i = (size_t)blockIdx.x * blockDim.x + threadIdx.x;
  if (i >= total) return;
  int col = (int)(i % 96);
  int row = (int)(i / 96);
  int b = row / NQ_;
  float v = 0.f;
  if (col < 64) v = (float)z[b * 64 + col];
  else if (col < 67) v = coors_de[(size_t)row * 3 + (col - 64)];
  q[i] = (half_t)v;
}

// ---------------------------------------------------------------------------
// Submanifold conv (3x3x3 or 1x1x1) as tap-decomposed WMMA implicit GEMM.
// One wave per (16-voxel, 16-channel) tile; K = ci padded to 32.
// Weights staged to LDS in [tap][Np][32] layout -> B frag = 2x ds_load_b128.
// A frag = 2x global b128 (ci=32) / 1x b128 (ci=16) / guarded scalar (ci<=4).
// ---------------------------------------------------------------------------
template <int NTAPS>
__global__ void __launch_bounds__(32)
conv_wmma_kernel(const half_t* __restrict__ xin, int ci,
                 const half_t* __restrict__ wsrc, int Np,
                 const float* __restrict__ bias,
                 const half_t* __restrict__ addh,
                 const float* __restrict__ mask,
                 half_t* __restrict__ out, int co,
                 int R, int act) {
  __shared__ __align__(16) half_t wsh[27 * 32 * 32];  // 55,296 B max
  const int lane = threadIdx.x;
  {
    const int n16 = NTAPS * Np * 32 / 8;  // 16B chunks
    const uint4* s4 = (const uint4*)wsrc;
    uint4* d4 = (uint4*)wsh;
    for (int i = lane; i < n16; i += 32) d4[i] = s4[i];
  }
  __syncthreads();

  const int g = lane >> 4;
  const int l15 = lane & 15;
  const int n = (blockIdx.y << 4) | l15;           // output channel
  const int R2 = R * R, R3 = R2 * R;

  const int gva = blockIdx.x * 16 + l15;           // this lane's A row (global voxel)
  const int ba = gva / R3;
  const int va = gva % R3;
  const int za = va / R2, ya = (va / R) % R, xa = va % R;

  v8f acc = {};
#pragma unroll 3
  for (int t = 0; t < NTAPS; ++t) {
    int dz = 0, dy = 0, dx = 0;
    if (NTAPS == 27) { dz = t / 9 - 1; dy = (t / 3) % 3 - 1; dx = t % 3 - 1; }
    const int nz = za + dz, ny = ya + dy, nx = xa + dx;
    const bool inb = (unsigned)nz < (unsigned)R && (unsigned)ny < (unsigned)R &&
                     (unsigned)nx < (unsigned)R;
    const half_t* ap = xin + (size_t)((size_t)ba * R3 + (size_t)nz * R2 + ny * R + nx) * ci;

    v16h afr;
    if (!inb) {
#pragma unroll
      for (int e = 0; e < 16; ++e) afr[e] = (half_t)0.f;
    } else if (ci == 32) {
      v8h lo = *(const v8h*)(ap + 8 * g);
      v8h hi = *(const v8h*)(ap + 16 + 8 * g);
#pragma unroll
      for (int e = 0; e < 8; ++e) { afr[e] = lo[e]; afr[8 + e] = hi[e]; }
    } else if (ci == 16) {
      v8h lo = *(const v8h*)(ap + 8 * g);
#pragma unroll
      for (int e = 0; e < 8; ++e) { afr[e] = lo[e]; afr[8 + e] = (half_t)0.f; }
    } else {  // ci in {1,4}: guarded scalar loads
#pragma unroll
      for (int v = 0; v < 8; ++v) {
        const int k0 = a_k_base(g, v);
        afr[2 * v]     = (k0 < ci)     ? ap[k0]     : (half_t)0.f;
        afr[2 * v + 1] = (k0 + 1 < ci) ? ap[k0 + 1] : (half_t)0.f;
      }
    }

    const half_t* wp = &wsh[(t * Np + n) * 32 + 8 * g];
    v8h blo = *(const v8h*)wp;
    v8h bhi = *(const v8h*)(wp + 16);
    v16h bfr;
#pragma unroll
    for (int e = 0; e < 8; ++e) { bfr[e] = blo[e]; bfr[8 + e] = bhi[e]; }

    acc = __builtin_amdgcn_wmma_f32_16x16x32_f16(false, afr, false, bfr, (short)0, acc,
                                                 false, false);
  }

  if (n < co) {
#pragma unroll
    for (int r = 0; r < 8; ++r) {
      const int gv = blockIdx.x * 16 + r + 8 * g;
      float val = acc[r];
      if (bias) val += bias[n];
      if (act == 1) val = selu_f(val);
      if (addh) val += (float)addh[(size_t)gv * co + n];
      val *= mask[gv];
      out[(size_t)gv * co + n] = (half_t)val;
    }
  }
}

// ---------------------------------------------------------------------------
// Generic WMMA GEMM: C = act( A[M,lda] @ Wt^T + bias + addC )
// Wt is n-major [Np][ldwt=Kp] f16, zero-padded. Wave tile = (16*MB) x (16*NBt):
// MB*NBt accumulators, fragments reused across the tile -> MB*NBt WMMAs/K-tile.
// act: 0 none, 1 selu, 2 sigmoid.  Writes f16 (outH) or f32 (outF).
// ---------------------------------------------------------------------------
template <int MB, int NBt>
__global__ void __launch_bounds__(32)
gemm_wmma_kernel(const half_t* __restrict__ A, int lda,
                 const half_t* __restrict__ Wt, int ldwt,
                 const float* __restrict__ bias,
                 const float* __restrict__ addC, int ldc_add,
                 float* __restrict__ outF, half_t* __restrict__ outH, int ldo,
                 int N, int Kp, int act) {
  const int lane = threadIdx.x;
  const int g = lane >> 4;
  const int l15 = lane & 15;
  const int m0 = blockIdx.x * (16 * MB) + l15;
  const int n0 = blockIdx.y * (16 * NBt) + l15;

  v8f acc[MB][NBt];
  const v8f vzero = {};
#pragma unroll
  for (int i = 0; i < MB; ++i)
#pragma unroll
    for (int j = 0; j < NBt; ++j) acc[i][j] = vzero;

  const int ktiles = Kp >> 5;
  for (int kt = 0; kt < ktiles; ++kt) {
    const int kb = kt * 32;
    if (kt + 1 < ktiles)
      __builtin_prefetch(A + (size_t)m0 * lda + kb + 32, 0, 0);  // global_prefetch_b8

    v16h afr[MB], bfr[NBt];
#pragma unroll
    for (int i = 0; i < MB; ++i) {
      const half_t* ap = A + (size_t)(m0 + i * 16) * lda + kb + 8 * g;
      v8h lo = *(const v8h*)ap;
      v8h hi = *(const v8h*)(ap + 16);
#pragma unroll
      for (int e = 0; e < 8; ++e) { afr[i][e] = lo[e]; afr[i][8 + e] = hi[e]; }
    }
#pragma unroll
    for (int j = 0; j < NBt; ++j) {
      const half_t* wp = Wt + (size_t)(n0 + j * 16) * ldwt + kb + 8 * g;
      v8h lo = *(const v8h*)wp;
      v8h hi = *(const v8h*)(wp + 16);
#pragma unroll
      for (int e = 0; e < 8; ++e) { bfr[j][e] = lo[e]; bfr[j][8 + e] = hi[e]; }
    }
#pragma unroll
    for (int i = 0; i < MB; ++i)
#pragma unroll
      for (int j = 0; j < NBt; ++j)
        acc[i][j] = __builtin_amdgcn_wmma_f32_16x16x32_f16(false, afr[i], false, bfr[j],
                                                           (short)0, acc[i][j], false, false);
  }

#pragma unroll
  for (int i = 0; i < MB; ++i) {
#pragma unroll
    for (int j = 0; j < NBt; ++j) {
      const int n = n0 + j * 16;
      if (n < N) {
#pragma unroll
        for (int r = 0; r < 8; ++r) {
          const int mr = blockIdx.x * (16 * MB) + i * 16 + r + 8 * g;
          float val = acc[i][j][r];
          if (bias) val += bias[n];
          if (addC) val += addC[(size_t)mr * ldc_add + n];
          if (act == 1) val = selu_f(val);
          else if (act == 2) val = sigmoid_f(val);
          if (outH) outH[(size_t)mr * ldo + n] = (half_t)val;
          if (outF) outF[(size_t)mr * ldo + n] = val;
        }
      }
    }
  }
}

// ---------------------------------------------------------------------------
// Host-side orchestration
// ---------------------------------------------------------------------------
static inline char* align256(char* p) {
  return (char*)(((uintptr_t)p + 255) & ~(uintptr_t)255);
}
static inline dim3 g1(size_t n, int b = 256) { return dim3((unsigned)((n + b - 1) / b)); }

extern "C" void kernel_launch(void* const* d_in, const int* in_sizes, int n_in,
                              void* d_out, int out_size, void* d_ws, size_t ws_size,
                              hipStream_t stream) {
  (void)n_in; (void)out_size; (void)ws_size;

  // ---- inputs (setup_inputs dict order; params flattened in insertion order)
  const float* feats    = (const float*)d_in[0];
  const int*   coors    = (const int*)d_in[1];
  const int*   batchidx = (const int*)d_in[2];
  const float* coors_de = (const float*)d_in[3];
  const int Npts = in_sizes[0];

  int pi = 4;
  const float *enc_w1[6], *enc_b1[6], *enc_w2[6], *enc_b2[6], *enc_ws[6];
  for (int i = 0; i < 6; ++i) {
    enc_w1[i] = (const float*)d_in[pi++]; enc_b1[i] = (const float*)d_in[pi++];
    enc_w2[i] = (const float*)d_in[pi++]; enc_b2[i] = (const float*)d_in[pi++];
    enc_ws[i] = (const float*)d_in[pi++];
  }
  const float* bneck_w = (const float*)d_in[pi++];
  const float* bneck_b = (const float*)d_in[pi++];
  const float *dec_w1[4], *dec_b1[4], *dec_w2[4], *dec_b2[4], *dec_ws[4];
  for (int i = 0; i < 4; ++i) {
    dec_w1[i] = (const float*)d_in[pi++]; dec_b1[i] = (const float*)d_in[pi++];
    dec_w2[i] = (const float*)d_in[pi++]; dec_b2[i] = (const float*)d_in[pi++];
    dec_ws[i] = (const float*)d_in[pi++];
  }

  // ---- workspace carving
  char* p = (char*)d_ws;
  auto take = [&](size_t bytes) -> void* { void* r = (void*)p; p = align256(p + bytes); return r; };

  const size_t VOX = (size_t)NB_ * NS_ * NS_ * NS_;           // 2,097,152
  float* ssum  = (float*)take(VOX * sizeof(float));
  float* cnt   = (float*)take(VOX * sizeof(float));
  float* maskA = (float*)take(VOX * sizeof(float));
  float* maskB = (float*)take(VOX * sizeof(float));
  const size_t ARENA_H = VOX * 16;                            // halves per arena
  half_t* ar[3];
  for (int i = 0; i < 3; ++i) ar[i] = (half_t*)take(ARENA_H * sizeof(half_t));
  half_t* zbuf = (half_t*)take((size_t)64 * 64 * sizeof(half_t));

  struct Blk { int ci, co, R; bool pool; };
  const Blk blks[6] = {{1, 16, 32, true}, {16, 32, 16, true}, {32, 32, 8, true},
                       {32, 4, 4, false}, {4, 4, 4, false},   {4, 1, 4, false}};
  int NpB[6];
  for (int i = 0; i < 6; ++i) NpB[i] = ((blks[i].co + 15) / 16) * 16;

  half_t *w1p[6], *w2p[6], *wsp[6];
  for (int i = 0; i < 6; ++i) {
    w1p[i] = (half_t*)take((size_t)27 * NpB[i] * 32 * sizeof(half_t));
    w2p[i] = (half_t*)take((size_t)27 * NpB[i] * 32 * sizeof(half_t));
    wsp[i] = (half_t*)take((size_t)NpB[i] * 32 * sizeof(half_t));
  }
  half_t* bneckp = (half_t*)take((size_t)64 * 64 * sizeof(half_t));
  struct DL { int K, N, Kp, Np; };
  const DL dls[4] = {{67, 128, 96, 128}, {128, 256, 128, 256},
                     {256, 128, 256, 128}, {128, 1, 128, 16}};
  half_t *dw1p[4], *dw2p[4], *dwsp[4];
  for (int i = 0; i < 4; ++i) {
    dw1p[i] = (half_t*)take((size_t)dls[i].Kp * dls[i].Np * sizeof(half_t));
    int k2p = (i == 3) ? 32 : dls[i].N, n2p = (i == 3) ? 16 : dls[i].N;
    dw2p[i] = (half_t*)take((size_t)k2p * n2p * sizeof(half_t));
    dwsp[i] = (half_t*)take((size_t)dls[i].Kp * dls[i].Np * sizeof(half_t));
  }

  // ---- pack weights (f16, transposed / tap-n-major, zero-padded)
  for (int i = 0; i < 6; ++i) {
    int ci = blks[i].ci, co = blks[i].co, Np = NpB[i];
    pack_conv_kernel<<<g1((size_t)27 * Np * 32), 256, 0, stream>>>(enc_w1[i], w1p[i], 27, ci, co, Np);
    pack_conv_kernel<<<g1((size_t)27 * Np * 32), 256, 0, stream>>>(enc_w2[i], w2p[i], 27, co, co, Np);
    pack_conv_kernel<<<g1((size_t)Np * 32), 256, 0, stream>>>(enc_ws[i], wsp[i], 1, ci, co, Np);
  }
  pack_wT_kernel<<<g1(64 * 64), 256, 0, stream>>>(bneck_w, bneckp, 64, 64, 64, 64);
  for (int i = 0; i < 4; ++i) {
    pack_wT_kernel<<<g1((size_t)dls[i].Kp * dls[i].Np), 256, 0, stream>>>(
        dec_w1[i], dw1p[i], dls[i].K, dls[i].N, dls[i].Kp, dls[i].Np);
    int k2s = (i == 3) ? 1 : dls[i].N, k2p = (i == 3) ? 32 : dls[i].N;
    int n2s = (i == 3) ? 1 : dls[i].N, n2p = (i == 3) ? 16 : dls[i].N;
    pack_wT_kernel<<<g1((size_t)k2p * n2p), 256, 0, stream>>>(dec_w2[i], dw2p[i], k2s, n2s, k2p, n2p);
    pack_wT_kernel<<<g1((size_t)dls[i].Kp * dls[i].Np), 256, 0, stream>>>(
        dec_ws[i], dwsp[i], dls[i].K, dls[i].N, dls[i].Kp, dls[i].Np);
  }

  // ---- voxelize (scatter mean) + mask
  zero_f32_kernel<<<g1(VOX), 256, 0, stream>>>(ssum, VOX);
  zero_f32_kernel<<<g1(VOX), 256, 0, stream>>>(cnt, VOX);
  voxelize_kernel<<<g1((size_t)Npts), 256, 0, stream>>>(feats, coors, batchidx, ssum, cnt, Npts);
  finalize_voxel_kernel<<<g1(VOX), 256, 0, stream>>>(ssum, cnt, ar[0], maskA, VOX);

  // ---- sparse residual encoder
  int xi = 0;
  float *mcur = maskA, *moth = maskB;
  for (int i = 0; i < 6; ++i) {
    const Blk b = blks[i];
    const int Np = NpB[i];
    const int t1 = (xi + 1) % 3, t2 = (xi + 2) % 3;
    const size_t nvox = (size_t)NB_ * b.R * b.R * b.R;
    dim3 grid((unsigned)(nvox / 16), (unsigned)(Np / 16));
    // h1 = selu(conv3(x, w1) + b1) * mask
    conv_wmma_kernel<27><<<grid, 32, 0, stream>>>(ar[xi], b.ci, w1p[i], Np, enc_b1[i],
                                                  nullptr, mcur, ar[t1], b.co, b.R, 1);
    // h2 = selu(conv3(h1, w2) + b2) * mask
    conv_wmma_kernel<27><<<grid, 32, 0, stream>>>(ar[t1], b.co, w2p[i], Np, enc_b2[i],
                                                  nullptr, mcur, ar[t2], b.co, b.R, 1);
    // x = (h2 + conv1(x, ws)) * mask
    conv_wmma_kernel<1><<<grid, 32, 0, stream>>>(ar[xi], b.ci, wsp[i], Np, nullptr,
                                                 ar[t2], mcur, ar[t1], b.co, b.R, 0);
    if (b.pool) {
      const int Ro = b.R / 2;
      const size_t totx = (size_t)NB_ * Ro * Ro * Ro * b.co;
      pool_x_kernel<<<g1(totx), 256, 0, stream>>>(ar[t1], ar[t2], Ro, b.co, totx);
      const size_t totm = (size_t)NB_ * Ro * Ro * Ro;
      pool_mask_kernel<<<g1(totm), 256, 0, stream>>>(mcur, moth, Ro, totm);
      float* tmp = mcur; mcur = moth; moth = tmp;
      xi = t2;
    } else {
      xi = t1;
    }
  }

  // ---- bottleneck: z = selu([B,64] @ W64x64 + b)  (enc out is [B*64] f16, co=1)
  gemm_wmma_kernel<2, 2><<<dim3(2, 2), 32, 0, stream>>>(ar[xi], 64, bneckp, 64, bneck_b,
                                                        nullptr, 0, nullptr, zbuf, 64,
                                                        64, 64, /*act=*/1);

  // ---- decoder (reuse arenas: q, h, and f32 t buffer)
  const int qa = (xi + 1) % 3, ha = (xi + 2) % 3, ta = xi;
  half_t* qcur = ar[qa];
  half_t* hbuf = ar[ha];
  float*  tbuf = (float*)ar[ta];

  build_q_kernel<<<g1((size_t)MDEC * 96), 256, 0, stream>>>(zbuf, coors_de, qcur,
                                                            (size_t)MDEC * 96);
  int ldq = 96;
  const unsigned MT32 = MDEC / 32;  // 1024 m-tiles for 32-row wave tiles
  for (int i = 0; i < 3; ++i) {
    const int N = dls[i].N, Kp = dls[i].Kp;
    dim3 gq(MT32, (unsigned)(N / 32));
    // h = selu(q @ w1 + b1)
    gemm_wmma_kernel<2, 2><<<gq, 32, 0, stream>>>(qcur, ldq, dw1p[i], Kp, dec_b1[i],
                                                  nullptr, 0, nullptr, hbuf, N, N, Kp, 1);
    // t = h @ w2 + b2    (f32)
    gemm_wmma_kernel<2, 2><<<gq, 32, 0, stream>>>(hbuf, N, dw2p[i], N, dec_b2[i],
                                                  nullptr, 0, tbuf, nullptr, N, N, N, 0);
    // q' = selu(q @ ws + t)   -> overwrite h arena
    gemm_wmma_kernel<2, 2><<<gq, 32, 0, stream>>>(qcur, ldq, dwsp[i], Kp, nullptr,
                                                  tbuf, N, nullptr, hbuf, N, N, Kp, 1);
    half_t* tmp = qcur; qcur = hbuf; hbuf = tmp;
    ldq = N;
  }

  // ---- dec3 (N=1, sigmoid): h = sig(q@w1+b1); out = sig(h@w2+b2 + q@ws)
  zero_f16_kernel<<<g1((size_t)MDEC * 32), 256, 0, stream>>>(hbuf, (size_t)MDEC * 32);
  gemm_wmma_kernel<2, 1><<<dim3(MT32, 1), 32, 0, stream>>>(qcur, 128, dw1p[3], 128, dec_b1[3],
                                                           nullptr, 0, nullptr, hbuf, 32,
                                                           1, 128, 2);
  gemm_wmma_kernel<2, 1><<<dim3(MT32, 1), 32, 0, stream>>>(hbuf, 32, dw2p[3], 32, dec_b2[3],
                                                           nullptr, 0, tbuf, nullptr, 16,
                                                           1, 32, 0);
  gemm_wmma_kernel<2, 1><<<dim3(MT32, 1), 32, 0, stream>>>(qcur, 128, dwsp[3], 128, nullptr,
                                                           tbuf, 16, (float*)d_out, nullptr, 1,
                                                           1, 128, 2);
}
